// L1TensorProduct_81844896793184
// MI455X (gfx1250) — matrix operand order repaired and checked by reading, not compiled, not run on hardware
//
#include <hip/hip_runtime.h>

typedef __attribute__((ext_vector_type(16))) __bf16 v16bf;
typedef __attribute__((ext_vector_type(8)))  __bf16 v8bf;
typedef __attribute__((ext_vector_type(8)))  float  v8f;
typedef __attribute__((ext_vector_type(4)))  float  v4f;

namespace {
constexpr float CG   = 0.5773502691896258f;   // 1/sqrt(3)
constexpr float CGC  = 0.4082482904638631f;   // 1/sqrt(6)
constexpr float NRM0 = 0.05103103630798288f;  // 1/sqrt(384)  (NORM0E == NORM0O)
constexpr float NRM1 = 0.07654655446197431f;  // sqrt(3/512)  (NORM1E == NORM1O)

constexpr int IN1D  = 1280;
constexpr int ROWS  = 16;        // batch rows per block
// bf16 transposed-weight workspace layout (element offsets)
constexpr int W0E_OFF = 0;       // [256][384]
constexpr int W0O_OFF = 98304;   // [256][384]
constexpr int W1E_OFF = 196608;  // [128][512]
constexpr int W1O_OFF = 262144;  // [128][512]
constexpr int W_TOTAL = 327680;
// padded LDS strides (in bf16 elements) to spread banks
constexpr int SAE = 392;         // 384 + 8
constexpr int SVS = 520;         // 512 + 8
} // namespace

// ---------------------------------------------------------------------------
// Pre-pass: W (f32, [K][N]) -> Wt (bf16, [N][K]) with NORM folded in.
// ---------------------------------------------------------------------------
__global__ void wconv_kernel(const float* __restrict__ W0e, const float* __restrict__ W0o,
                             const float* __restrict__ W1e, const float* __restrict__ W1o,
                             __bf16* __restrict__ wt) {
  int idx = blockIdx.x * blockDim.x + threadIdx.x;
  if (idx >= W_TOTAL) return;
  float val;
  if (idx < W0O_OFF) {
    int i = idx;           int n = i / 384, k = i % 384;
    val = W0e[k * 256 + n] * NRM0;
  } else if (idx < W1E_OFF) {
    int i = idx - W0O_OFF; int n = i / 384, k = i % 384;
    val = W0o[k * 256 + n] * NRM0;
  } else if (idx < W1O_OFF) {
    int i = idx - W1E_OFF; int n = i / 512, k = i % 512;
    val = W1e[k * 128 + n] * NRM1;
  } else {
    int i = idx - W1O_OFF; int n = i / 512, k = i % 512;
    val = W1o[k * 128 + n] * NRM1;
  }
  wt[idx] = (__bf16)val;
}

// ---------------------------------------------------------------------------
// 16x32 tile-pair: shared A fragment, two B columns-of-16, K even/odd split
// => 4 independent WMMA chains. Partial unroll keeps VGPRs < 256.
// ---------------------------------------------------------------------------
template<int K>
__device__ __forceinline__ void gemm_pair(const __bf16* arow,
                                          const __bf16* w0, const __bf16* w1,
                                          float* op0, float* op1) {
  v8f a00 = {}, a10 = {}, a01 = {}, a11 = {};
#pragma unroll 2
  for (int k = 0; k < K; k += 64) {
    union { v16bf v; v8bf h[2]; } fa0, fa1;
    fa0.h[0] = *(const v8bf*)(arow + k);
    fa0.h[1] = *(const v8bf*)(arow + k + 8);
    fa1.h[0] = *(const v8bf*)(arow + k + 32);
    fa1.h[1] = *(const v8bf*)(arow + k + 40);
    v16bf b00 = *(const v16bf*)(w0 + k);
    v16bf b10 = *(const v16bf*)(w1 + k);
    v16bf b01 = *(const v16bf*)(w0 + k + 32);
    v16bf b11 = *(const v16bf*)(w1 + k + 32);
    a00 = __builtin_amdgcn_wmma_f32_16x16x32_bf16(false, fa0.v, false, b00, (short)0, a00, false, false);
    a10 = __builtin_amdgcn_wmma_f32_16x16x32_bf16(false, fa0.v, false, b10, (short)0, a10, false, false);
    a01 = __builtin_amdgcn_wmma_f32_16x16x32_bf16(false, fa1.v, false, b01, (short)0, a01, false, false);
    a11 = __builtin_amdgcn_wmma_f32_16x16x32_bf16(false, fa1.v, false, b11, (short)0, a11, false, false);
  }
  v8f r0 = a00 + a01;
  v8f r1 = a10 + a11;
#pragma unroll
  for (int r = 0; r < 8; ++r) {
    __builtin_nontemporal_store(r0[r], op0 + (size_t)r * IN1D);
    __builtin_nontemporal_store(r1[r], op1 + (size_t)r * IN1D);
  }
}

// ---------------------------------------------------------------------------
// 16x16x3-component triple: the 3 spatial components share the SAME weight
// columns, so one B fragment feeds 3 independent WMMA chains (3x less L2
// traffic), and the 3 outputs are consecutive dwords (contiguous stores).
// ---------------------------------------------------------------------------
template<int K>
__device__ __forceinline__ void gemm_triple(const __bf16* a0row, const __bf16* a1row,
                                            const __bf16* a2row, const __bf16* w,
                                            float* op) {
  v8f c0 = {}, c1 = {}, c2 = {};
#pragma unroll 4
  for (int k = 0; k < K; k += 32) {
    union { v16bf v; v8bf h[2]; } f0, f1, f2;
    f0.h[0] = *(const v8bf*)(a0row + k);  f0.h[1] = *(const v8bf*)(a0row + k + 8);
    f1.h[0] = *(const v8bf*)(a1row + k);  f1.h[1] = *(const v8bf*)(a1row + k + 8);
    f2.h[0] = *(const v8bf*)(a2row + k);  f2.h[1] = *(const v8bf*)(a2row + k + 8);
    v16bf b = *(const v16bf*)(w + k);
    c0 = __builtin_amdgcn_wmma_f32_16x16x32_bf16(false, f0.v, false, b, (short)0, c0, false, false);
    c1 = __builtin_amdgcn_wmma_f32_16x16x32_bf16(false, f1.v, false, b, (short)0, c1, false, false);
    c2 = __builtin_amdgcn_wmma_f32_16x16x32_bf16(false, f2.v, false, b, (short)0, c2, false, false);
  }
#pragma unroll
  for (int r = 0; r < 8; ++r) {
    float* p = op + (size_t)r * IN1D;
    __builtin_nontemporal_store(c0[r], p + 0);
    __builtin_nontemporal_store(c1[r], p + 1);
    __builtin_nontemporal_store(c2[r], p + 2);
  }
}

// ---------------------------------------------------------------------------
// Fused kernel: bf16 activations in LDS, then branch-free WMMA work per wave.
// ---------------------------------------------------------------------------
__global__ __launch_bounds__(256)
void l1tp_kernel(const float* __restrict__ in1, const float* __restrict__ in2,
                 const __bf16* __restrict__ wt, float* __restrict__ out) {
  __shared__ __bf16 sAE[ROWS * SAE];       // f0e  rows
  __shared__ __bf16 sAO[ROWS * SAE];       // f0o  rows
  __shared__ __bf16 sVE3[3][ROWS * SVS];   // vecs_e per component
  __shared__ __bf16 sVO3[3][ROWS * SVS];   // vecs_o per component

  const int tid = threadIdx.x;
  const int r0  = blockIdx.x * ROWS;

  // ---- Phase 1: activation build (16 threads per row, b128 NT loads) ----
  {
    const int row = tid >> 4;
    const int sub = tid & 15;
    const float* i2 = in2 + (size_t)(r0 + row) * 4;
    const float s = i2[0], v0 = i2[1], v1 = i2[2], v2 = i2[3];
    const v4f* inr4 = (const v4f*)(in1 + (size_t)(r0 + row) * IN1D);

    __bf16* aE  = sAE     + row * SAE;
    __bf16* aO  = sAO     + row * SAE;
    __bf16* vE0 = sVE3[0] + row * SVS;
    __bf16* vE1 = sVE3[1] + row * SVS;
    __bf16* vE2 = sVE3[2] + row * SVS;
    __bf16* vO0 = sVO3[0] + row * SVS;
    __bf16* vO1 = sVO3[1] + row * SVS;
    __bf16* vO2 = sVO3[2] + row * SVS;

#pragma unroll
    for (int i = 0; i < 4; ++i) {          // x0e (256) + x0o (256): 64B/lane
      v4f qe = __builtin_nontemporal_load(inr4 + sub * 4 + i);
      v4f qo = __builtin_nontemporal_load(inr4 + 64 + sub * 4 + i);
#pragma unroll
      for (int c = 0; c < 4; ++c) {
        int j = sub * 16 + i * 4 + c;
        float xe = qe[c], xo = qo[c];
        aE[j] = (__bf16)(xe * s);
        aO[j] = (__bf16)(xo * s);
        float ce = CG * xe, co = CG * xo;
        vO0[j] = (__bf16)(ce * v0); vO1[j] = (__bf16)(ce * v1); vO2[j] = (__bf16)(ce * v2);
        vE0[j] = (__bf16)(co * v0); vE1[j] = (__bf16)(co * v1); vE2[j] = (__bf16)(co * v2);
      }
    }
    float be[24], bo[24];                  // x1e / x1o: 6 x b128 per lane each
#pragma unroll
    for (int i = 0; i < 6; ++i) {
      v4f qe = __builtin_nontemporal_load(inr4 + 128 + sub * 6 + i);
      v4f qo = __builtin_nontemporal_load(inr4 + 224 + sub * 6 + i);
#pragma unroll
      for (int c = 0; c < 4; ++c) { be[i * 4 + c] = qe[c]; bo[i * 4 + c] = qo[c]; }
    }
#pragma unroll
    for (int t = 0; t < 8; ++t) {
      int j = sub * 8 + t;
      float e0 = be[3 * t], e1 = be[3 * t + 1], e2 = be[3 * t + 2];
      float o0 = bo[3 * t], o1 = bo[3 * t + 1], o2 = bo[3 * t + 2];
      aO[256 + j] = (__bf16)(CG * (e0 * v0 + e1 * v1 + e2 * v2));
      aE[256 + j] = (__bf16)(CG * (o0 * v0 + o1 * v1 + o2 * v2));
      float cs = CG * s;
      vE0[256 + j] = (__bf16)(cs * e0); vE1[256 + j] = (__bf16)(cs * e1); vE2[256 + j] = (__bf16)(cs * e2);
      vO0[256 + j] = (__bf16)(cs * o0); vO1[256 + j] = (__bf16)(cs * o1); vO2[256 + j] = (__bf16)(cs * o2);
      vE0[384 + j] = (__bf16)(CGC * (o1 * v2 - o2 * v1));   // CGC*cross(x1o, v)
      vE1[384 + j] = (__bf16)(CGC * (o2 * v0 - o0 * v2));
      vE2[384 + j] = (__bf16)(CGC * (o0 * v1 - o1 * v0));
      vO0[384 + j] = (__bf16)(CGC * (e1 * v2 - e2 * v1));   // CGC*cross(x1e, v)
      vO1[384 + j] = (__bf16)(CGC * (e2 * v0 - e0 * v2));
      vO2[384 + j] = (__bf16)(CGC * (e0 * v1 - e1 * v0));
    }
  }
  __syncthreads();

  // ---- Phase 2: branch-free. Each wave: one pair from GEMM1 & GEMM2,
  //      one component-triple from GEMM3 & GEMM4 (144 WMMAs each). ----
  const int lane = tid & 31;
  const int wave = tid >> 5;
  const int m    = lane & 15;   // row (A half) / column (B,C,D)
  const int hi   = lane >> 4;   // which K-half this lane holds
  float* outRow  = out + (size_t)(r0 + hi * 8) * IN1D;
  const int aoff = m * SAE + hi * 16;
  const int voff = m * SVS + hi * 16;

  { // GEMM1: out0e, K=384, pair of 16-col tiles
    const int n0 = wave * 32;
    const __bf16* w0 = wt + W0E_OFF + (size_t)(n0 + m) * 384 + hi * 16;
    gemm_pair<384>(sAE + aoff, w0, w0 + 16 * 384,
                   outRow + (n0 + m), outRow + (n0 + 16 + m));
  }
  { // GEMM2: out0o, K=384
    const int n0 = wave * 32;
    const __bf16* w0 = wt + W0O_OFF + (size_t)(n0 + m) * 384 + hi * 16;
    gemm_pair<384>(sAO + aoff, w0, w0 + 16 * 384,
                   outRow + 256 + (n0 + m), outRow + 256 + (n0 + 16 + m));
  }
  { // GEMM3: out1e, K=512, 3 components share B
    const int n0 = wave * 16;
    const __bf16* w = wt + W1E_OFF + (size_t)(n0 + m) * 512 + hi * 16;
    gemm_triple<512>(sVE3[0] + voff, sVE3[1] + voff, sVE3[2] + voff, w,
                     outRow + 512 + (n0 + m) * 3);
  }
  { // GEMM4: out1o, K=512
    const int n0 = wave * 16;
    const __bf16* w = wt + W1O_OFF + (size_t)(n0 + m) * 512 + hi * 16;
    gemm_triple<512>(sVO3[0] + voff, sVO3[1] + voff, sVO3[2] + voff, w,
                     outRow + 896 + (n0 + m) * 3);
  }
}

// ---------------------------------------------------------------------------
extern "C" void kernel_launch(void* const* d_in, const int* in_sizes, int n_in,
                              void* d_out, int out_size, void* d_ws, size_t ws_size,
                              hipStream_t stream) {
  const float* in1 = (const float*)d_in[0];
  const float* in2 = (const float*)d_in[1];
  const float* W0e = (const float*)d_in[2];
  const float* W0o = (const float*)d_in[3];
  const float* W1e = (const float*)d_in[4];
  const float* W1o = (const float*)d_in[5];
  __bf16* wt = (__bf16*)d_ws;              // 327680 bf16 = 640 KB scratch
  float* out = (float*)d_out;

  const int B = in_sizes[1] / 4;           // 65536

  wconv_kernel<<<(W_TOTAL + 255) / 256, 256, 0, stream>>>(W0e, W0o, W1e, W1o, wt);
  l1tp_kernel<<<B / ROWS, 256, 0, stream>>>(in1, in2, wt, out);
}
